// GCN_24395414241281
// MI455X (gfx1250) — compile-verified, matrix operand order checked by reading, not run on hardware
//
#include <hip/hip_runtime.h>
#include <math.h>

typedef __attribute__((ext_vector_type(16))) _Float16 v16h;
typedef __attribute__((ext_vector_type(8)))  float    v8f;

#define HDIM  128
#define INDIM 64

// Fast reciprocal: single v_rcp_f32 (~1 ulp), avoids IEEE div expansion.
__device__ __forceinline__ float frcp(float x) { return __builtin_amdgcn_rcpf(x); }

// mish(x) = x * tanh(softplus(x)) = x * (t^2-1)/(t^2+1), t = 1+exp(x).
// Clamp exp arg at 20: for x>=20 the ratio is 1 to fp32 precision (result = x),
// for x->-inf it tends to 0. One v_exp_f32 + one v_rcp_f32 + a few VALU ops.
__device__ __forceinline__ float mishf(float x) {
    const float e  = __expf(fminf(x, 20.0f));
    const float t  = 1.0f + e;
    const float t2 = t * t;
    return x * (t2 - 1.0f) * frcp(t2 + 1.0f);
}

// ---------------------------------------------------------------------------
// Dense GEMM: out[M,N] = act( A[M,K] @ W[N,K]^T + bias[N] )
// One wave (32 threads) per 16x16 output tile, v_wmma_f32_16x16x32_f16.
// K templated (64/128) -> fully unrolled chain of 2-4 WMMAs; ACT templated.
// M % 16 == 0, N % 16 == 0 (holds: M=40000, N in {64,128}).
// ACT: 0 = linear, 1 = relu, 2 = mish
// ---------------------------------------------------------------------------
template<int K, int ACT>
__global__ void gemm_wmma_kernel(const float* __restrict__ A,
                                 const float* __restrict__ W,
                                 const float* __restrict__ bias,
                                 float* __restrict__ out,
                                 int N)
{
    const int lane    = threadIdx.x & 31;
    const int rowBase = blockIdx.x * 16;
    const int colBase = blockIdx.y * 16;
    const int l15     = lane & 15;
    const bool hi     = lane >= 16;
    const int kSub    = hi ? 8 : 0;

    const float* ap = A + (size_t)(rowBase + l15) * K + kSub;  // A-fragment row
    const float* wp = W + (size_t)(colBase + l15) * K + kSub;  // B-fragment column

    v8f acc = {};
    #pragma unroll
    for (int k0 = 0; k0 < K; k0 += 32) {
        const float4* ap4 = (const float4*)(ap + k0);  // 16B aligned (kSub mult of 8)
        const float4* wp4 = (const float4*)(wp + k0);
        float4 a0 = ap4[0], a1 = ap4[1], a2 = ap4[4], a3 = ap4[5]; // K: +0..7, +16..23
        float4 b0 = wp4[0], b1 = wp4[1], b2 = wp4[4], b3 = wp4[5];

        float af[16] = { a0.x,a0.y,a0.z,a0.w, a1.x,a1.y,a1.z,a1.w,
                         a2.x,a2.y,a2.z,a2.w, a3.x,a3.y,a3.z,a3.w };
        float bf[16] = { b0.x,b0.y,b0.z,b0.w, b1.x,b1.y,b1.z,b1.w,
                         b2.x,b2.y,b2.z,b2.w, b3.x,b3.y,b3.z,b3.w };
        v16h a, b;
        #pragma unroll
        for (int i = 0; i < 16; ++i) { a[i] = (_Float16)af[i]; b[i] = (_Float16)bf[i]; }

        acc = __builtin_amdgcn_wmma_f32_16x16x32_f16(
            /*neg_a=*/false, a, /*neg_b=*/false, b,
            /*c_mod=*/(short)0, acc, /*reuse_a=*/false, /*reuse_b=*/false);
    }

    const int rowOut = rowBase + (hi ? 8 : 0);
    const int nB     = colBase + l15;
    const float bval = bias[nB];
    #pragma unroll
    for (int v = 0; v < 8; ++v) {
        float val = acc[v] + bval;
        if (ACT == 1)      val = fmaxf(val, 0.0f);
        else if (ACT == 2) val = mishf(val);
        out[(size_t)(rowOut + v) * N + nB] = val;
    }
}

// ---------------------------------------------------------------------------
// Embedding overwrite for nodes >= n_hea, then row-wise L2 normalize.
// One wave per node; lane handles 4 contiguous features (32*4 = 128 = H).
// ---------------------------------------------------------------------------
__global__ void embed_l2norm_kernel(float* __restrict__ h,
                                    const float* __restrict__ emb_table,
                                    const int* __restrict__ x_id,
                                    const int* __restrict__ dict_table,
                                    const int* __restrict__ n_hea_p,
                                    int N)
{
    const int wid  = threadIdx.x >> 5;
    const int lane = threadIdx.x & 31;
    const int i    = blockIdx.x * (blockDim.x >> 5) + wid;
    if (i >= N) return;
    const int n_hea = n_hea_p[0];
    const int f = lane * 4;

    float4 v;
    if (i >= n_hea) {
        const int id = dict_table[x_id[i]];
        v = *(const float4*)(emb_table + (size_t)id * HDIM + f);
    } else {
        v = *(const float4*)(h + (size_t)i * HDIM + f);
    }
    float ss = v.x*v.x + v.y*v.y + v.z*v.z + v.w*v.w;
    #pragma unroll
    for (int off = 16; off > 0; off >>= 1) ss += __shfl_xor(ss, off);
    const float inv = frcp(fmaxf(sqrtf(ss), 1e-12f));
    float4 o; o.x = v.x*inv; o.y = v.y*inv; o.z = v.z*inv; o.w = v.w*inv;
    *(float4*)(h + (size_t)i * HDIM + f) = o;
}

// ---------------------------------------------------------------------------
// Per-edge pass: te = h[row]+rel; seg[col] += te; cnt[col] += 1;
// pos = ||h[row]+rel-h[col]||; neg = ||h[neg]+rel-h[col]||.
// One wave per edge, 4 features per lane.
// ---------------------------------------------------------------------------
__global__ void edge_kernel(const float* __restrict__ h,
                            const float* __restrict__ emb_edge,
                            const int* __restrict__ row, const int* __restrict__ col,
                            const int* __restrict__ eattr, const int* __restrict__ negi,
                            float* __restrict__ seg, float* __restrict__ cnt,
                            float* __restrict__ pos_out, float* __restrict__ neg_out,
                            int E)
{
    const int wid  = threadIdx.x >> 5;
    const int lane = threadIdx.x & 31;
    const int e    = blockIdx.x * (blockDim.x >> 5) + wid;
    if (e >= E) return;

    const int r = row[e], c = col[e], t = eattr[e], g = negi[e];
    const int f = lane * 4;

    const float4 hr = *(const float4*)(h + (size_t)r * HDIM + f);
    const float4 rl = *(const float4*)(emb_edge + (size_t)t * HDIM + f);
    const float4 hc = *(const float4*)(h + (size_t)c * HDIM + f);
    const float4 hg = *(const float4*)(h + (size_t)g * HDIM + f);

    float4 te; te.x = hr.x+rl.x; te.y = hr.y+rl.y; te.z = hr.z+rl.z; te.w = hr.w+rl.w;

    float* sp = seg + (size_t)c * HDIM + f;
    atomicAdd(sp + 0, te.x); atomicAdd(sp + 1, te.y);
    atomicAdd(sp + 2, te.z); atomicAdd(sp + 3, te.w);

    float dx = te.x - hc.x, dy = te.y - hc.y, dz = te.z - hc.z, dw = te.w - hc.w;
    float pos = dx*dx + dy*dy + dz*dz + dw*dw;
    dx = hg.x + rl.x - hc.x; dy = hg.y + rl.y - hc.y;
    dz = hg.z + rl.z - hc.z; dw = hg.w + rl.w - hc.w;
    float neg = dx*dx + dy*dy + dz*dz + dw*dw;

    #pragma unroll
    for (int off = 16; off > 0; off >>= 1) {
        pos += __shfl_down(pos, off);
        neg += __shfl_down(neg, off);
    }
    if (lane == 0) {
        pos_out[e] = sqrtf(pos);
        neg_out[e] = sqrtf(neg);
        atomicAdd(cnt + c, 1.0f);
    }
}

// ---------------------------------------------------------------------------
// score_semantic[i] = || h[i] - seg[i]/cnt[i] ||  for i < n_hea. Wave per node.
// ---------------------------------------------------------------------------
__global__ void semantic_kernel(const float* __restrict__ h,
                                const float* __restrict__ seg,
                                const float* __restrict__ cnt,
                                float* __restrict__ sem_out, int n_hea)
{
    const int wid  = threadIdx.x >> 5;
    const int lane = threadIdx.x & 31;
    const int i    = blockIdx.x * (blockDim.x >> 5) + wid;
    if (i >= n_hea) return;
    const int f = lane * 4;
    const float icnt = frcp(cnt[i]);
    const float4 s  = *(const float4*)(seg + (size_t)i * HDIM + f);
    const float4 hh = *(const float4*)(h   + (size_t)i * HDIM + f);
    float dx = hh.x - s.x*icnt, dy = hh.y - s.y*icnt,
          dz = hh.z - s.z*icnt, dw = hh.w - s.w*icnt;
    float acc = dx*dx + dy*dy + dz*dz + dw*dw;
    #pragma unroll
    for (int off = 16; off > 0; off >>= 1) acc += __shfl_down(acc, off);
    if (lane == 0) sem_out[i] = sqrtf(acc);
}

// ---------------------------------------------------------------------------
// GCN accumulator init with fused self-loop: acc[i,:] = lin[i,:] / (cnt[i]+1)
// (dis[i]^2 = 1/deg[i], deg = cnt_col + 1 from self-loops). Thread per element.
// ---------------------------------------------------------------------------
__global__ void gcn_init_kernel(const float* __restrict__ lin,
                                const float* __restrict__ cnt,
                                float* __restrict__ acc, size_t total)
{
    const size_t idx = (size_t)blockIdx.x * blockDim.x + threadIdx.x;
    if (idx >= total) return;
    const int i = (int)(idx >> 7);              // HDIM == 128
    const float inv = frcp(cnt[i] + 1.0f);      // dis[i]^2
    acc[idx] = inv * lin[idx];
}

// ---------------------------------------------------------------------------
// GCN scatter: acc[c,:] += dis[r]*dis[c] * lin[r,:]. Wave per edge.
// ---------------------------------------------------------------------------
__global__ void gcn_scatter_kernel(const float* __restrict__ lin,
                                   const int* __restrict__ row,
                                   const int* __restrict__ col,
                                   const float* __restrict__ cnt,
                                   float* __restrict__ acc, int E)
{
    const int wid  = threadIdx.x >> 5;
    const int lane = threadIdx.x & 31;
    const int e    = blockIdx.x * (blockDim.x >> 5) + wid;
    if (e >= E) return;
    const int r = row[e], c = col[e];
    const float nrm = rsqrtf(cnt[r] + 1.0f) * rsqrtf(cnt[c] + 1.0f);
    const int f = lane * 4;
    const float4 v = *(const float4*)(lin + (size_t)r * HDIM + f);
    float* ap = acc + (size_t)c * HDIM + f;
    atomicAdd(ap + 0, nrm * v.x); atomicAdd(ap + 1, nrm * v.y);
    atomicAdd(ap + 2, nrm * v.z); atomicAdd(ap + 3, nrm * v.w);
}

__global__ void relu_kernel(float* __restrict__ x, size_t total)
{
    const size_t idx = (size_t)blockIdx.x * blockDim.x + threadIdx.x;
    if (idx < total) x[idx] = fmaxf(x[idx], 0.0f);
}

// out = relu(a) + x1  (fused outer relu of conv2 + residual)
__global__ void relu_add_kernel(const float* __restrict__ a,
                                const float* __restrict__ x1,
                                float* __restrict__ out, size_t total)
{
    const size_t idx = (size_t)blockIdx.x * blockDim.x + threadIdx.x;
    if (idx < total) out[idx] = fmaxf(a[idx], 0.0f) + x1[idx];
}

// ---------------------------------------------------------------------------
// Head: out[i] = sigmoid( dot(t[i,0:64], l2w) + l2b ). Wave per node, 2 feats/lane.
// ---------------------------------------------------------------------------
__global__ void head_kernel(const float* __restrict__ t,
                            const float* __restrict__ l2w,
                            const float* __restrict__ l2b,
                            float* __restrict__ out, int N)
{
    const int wid  = threadIdx.x >> 5;
    const int lane = threadIdx.x & 31;
    const int i    = blockIdx.x * (blockDim.x >> 5) + wid;
    if (i >= N) return;
    const float* ti = t + (size_t)i * 64;
    float acc = ti[lane] * l2w[lane] + ti[lane + 32] * l2w[lane + 32];
    #pragma unroll
    for (int off = 16; off > 0; off >>= 1) acc += __shfl_down(acc, off);
    if (lane == 0) out[i] = frcp(1.0f + __expf(-(acc + l2b[0])));
}

// ---------------------------------------------------------------------------
extern "C" void kernel_launch(void* const* d_in, const int* in_sizes, int n_in,
                              void* d_out, int out_size, void* d_ws, size_t ws_size,
                              hipStream_t stream)
{
    (void)n_in; (void)ws_size;

    const float* x          = (const float*)d_in[0];
    /* d_in[1] edge_weight intentionally unused (matches reference) */
    const float* ew1 = (const float*)d_in[2];  const float* eb1 = (const float*)d_in[3];
    const float* ew2 = (const float*)d_in[4];  const float* eb2 = (const float*)d_in[5];
    const float* ew3 = (const float*)d_in[6];  const float* eb3 = (const float*)d_in[7];
    const float* emb_table = (const float*)d_in[8];
    const float* emb_edge  = (const float*)d_in[9];
    const float* c1w = (const float*)d_in[10]; const float* c1b = (const float*)d_in[11];
    const float* c2w = (const float*)d_in[12]; const float* c2b = (const float*)d_in[13];
    const float* l1w = (const float*)d_in[14]; const float* l1b = (const float*)d_in[15];
    const float* l2w = (const float*)d_in[16]; const float* l2b = (const float*)d_in[17];
    const int* x_id       = (const int*)d_in[18];
    const int* dict_table = (const int*)d_in[19];
    const int* row        = (const int*)d_in[20];
    const int* col        = (const int*)d_in[21];
    const int* eattr      = (const int*)d_in[22];
    const int* negi       = (const int*)d_in[23];
    const int* n_hea_p    = (const int*)d_in[24];

    const int N = in_sizes[0] / INDIM;           // 40000
    const int E = in_sizes[20];                  // 640000
    const int n_hea = out_size - N - 2 * E;      // 10000 (host-derivable)

    float* out_main = (float*)d_out;             // [N]
    float* pos_out  = out_main + N;              // [E]
    float* neg_out  = pos_out + E;               // [E]
    float* sem_out  = neg_out + E;               // [n_hea]

    float* ws  = (float*)d_ws;
    float* t1  = ws;                              // N*64  (enc1 out, later l1 out)
    float* h   = t1 + (size_t)N * 64;             // N*128 (encoder out = x1, preserved)
    float* B   = h  + (size_t)N * HDIM;           // N*128 (GEMM outputs)
    float* C   = B  + (size_t)N * HDIM;           // N*128 (t2 / seg / conv1 acc)
    float* D   = C  + (size_t)N * HDIM;           // N*128 (conv2 acc)
    float* cnt = D  + (size_t)N * HDIM;           // N

    const size_t tot = (size_t)N * HDIM;
    const dim3 blk32(32, 1, 1);
    const int wavesPerBlk = 8;                    // 256-thread blocks, wave-per-item kernels

    // --- encoder MLP (WMMA) ---
    gemm_wmma_kernel<64, 2><<<dim3(N/16, 64/16),  blk32, 0, stream>>>(x,  ew1, eb1, t1, 64);
    gemm_wmma_kernel<64, 2><<<dim3(N/16, 128/16), blk32, 0, stream>>>(t1, ew2, eb2, C,  128);
    gemm_wmma_kernel<128,0><<<dim3(N/16, 128/16), blk32, 0, stream>>>(C,  ew3, eb3, h,  128);

    // --- embedding overwrite + L2 norm ---
    embed_l2norm_kernel<<<(N + wavesPerBlk - 1)/wavesPerBlk, 256, 0, stream>>>(
        h, emb_table, x_id, dict_table, n_hea_p, N);

    // --- per-edge scores + segment sums ---
    hipMemsetAsync(C,   0, tot * sizeof(float), stream);
    hipMemsetAsync(cnt, 0, (size_t)N * sizeof(float), stream);
    edge_kernel<<<(E + wavesPerBlk - 1)/wavesPerBlk, 256, 0, stream>>>(
        h, emb_edge, row, col, eattr, negi, C, cnt, pos_out, neg_out, E);
    semantic_kernel<<<(n_hea + wavesPerBlk - 1)/wavesPerBlk, 256, 0, stream>>>(
        h, C, cnt, sem_out, n_hea);

    // --- GCN conv 1: lin(WMMA,relu) -> self-loop init -> scatter -> outer relu ---
    gemm_wmma_kernel<128,1><<<dim3(N/16, 128/16), blk32, 0, stream>>>(h, c1w, c1b, B, 128);
    gcn_init_kernel<<<(unsigned)((tot + 255)/256), 256, 0, stream>>>(B, cnt, C, tot);
    gcn_scatter_kernel<<<(E + wavesPerBlk - 1)/wavesPerBlk, 256, 0, stream>>>(B, row, col, cnt, C, E);
    relu_kernel<<<(unsigned)((tot + 255)/256), 256, 0, stream>>>(C, tot);

    // --- GCN conv 2 + residual (outer relu fused into residual add) ---
    gemm_wmma_kernel<128,1><<<dim3(N/16, 128/16), blk32, 0, stream>>>(C, c2w, c2b, B, 128);
    gcn_init_kernel<<<(unsigned)((tot + 255)/256), 256, 0, stream>>>(B, cnt, D, tot);
    gcn_scatter_kernel<<<(E + wavesPerBlk - 1)/wavesPerBlk, 256, 0, stream>>>(B, row, col, cnt, D, E);
    relu_add_kernel<<<(unsigned)((tot + 255)/256), 256, 0, stream>>>(D, h, B, tot);

    // --- head MLP: 128->64 (WMMA, relu) then 64->1 sigmoid ---
    gemm_wmma_kernel<128,1><<<dim3(N/16, 64/16), blk32, 0, stream>>>(B, l1w, l1b, t1, 64);
    head_kernel<<<(N + wavesPerBlk - 1)/wavesPerBlk, 256, 0, stream>>>(t1, l2w, l2b, out_main, N);
}
